// MoELayer_21835613733541
// MI455X (gfx1250) — matrix operand order repaired and checked by reading, not compiled, not run on hardware
//
#include <hip/hip_runtime.h>

// ---------------- problem constants (from reference) ----------------
#define TT   4096        // tokens = B*S
#define DM   1024        // d_model
#define FF   2048        // d_ff
#define NE   8           // experts
#define TOPK 2

// LDS padded strides (bf16 units) to break bank conflicts on A-fragment loads
#define XS 1032          // 1024 + 8
#define YS 136           // 128 + 8

typedef __bf16 bf16_t;
typedef __attribute__((ext_vector_type(16))) __bf16 v16bf;
typedef __attribute__((ext_vector_type(8)))  __bf16 v8bf;
typedef __attribute__((ext_vector_type(8)))  float  v8f;

static __device__ inline unsigned short f32_to_bf16_rne(float f) {
    unsigned int u = __float_as_uint(f);
    unsigned int lsb = (u >> 16) & 1u;
    u += 0x7fffu + lsb;
    return (unsigned short)(u >> 16);
}

static __device__ inline v16bf cat16(v8bf lo, v8bf hi) {
    v16bf r;
#pragma unroll
    for (int i = 0; i < 8; ++i) { r[i] = lo[i]; r[i + 8] = hi[i]; }
    return r;
}

// ---------------- fp32 -> bf16 bulk convert ----------------
__global__ void k_f32_to_bf16(const float* __restrict__ src,
                              unsigned short* __restrict__ dst, int n) {
    int i = blockIdx.x * blockDim.x + threadIdx.x;
    int stride = gridDim.x * blockDim.x;
    for (; i < n; i += stride) dst[i] = f32_to_bf16_rne(src[i]);
}

// ---------------- router: logits, softmax probs, top-2 scatter ----------------
__global__ __launch_bounds__(256) void k_router(
    const float* __restrict__ x, const float* __restrict__ gate_w,
    float* __restrict__ probs, int* __restrict__ toklist,
    float* __restrict__ wlist, int* __restrict__ count) {
    __shared__ float red[NE][256];
    const int t = blockIdx.x;
    const int tid = threadIdx.x;
    const float* xr = x + (size_t)t * DM;

    float p[NE];
#pragma unroll
    for (int e = 0; e < NE; ++e) p[e] = 0.f;
    for (int j = tid; j < DM; j += 256) {
        float xv = xr[j];
#pragma unroll
        for (int e = 0; e < NE; ++e) p[e] += xv * gate_w[e * DM + j];
    }
#pragma unroll
    for (int e = 0; e < NE; ++e) red[e][tid] = p[e];
    __syncthreads();
    for (int off = 128; off > 0; off >>= 1) {
        if (tid < off) {
#pragma unroll
            for (int e = 0; e < NE; ++e) red[e][tid] += red[e][tid + off];
        }
        __syncthreads();
    }
    if (tid == 0) {
        float lg[NE];
#pragma unroll
        for (int e = 0; e < NE; ++e) lg[e] = red[e][0];
        // full softmax (for aux loss)
        float mx = lg[0];
#pragma unroll
        for (int e = 1; e < NE; ++e) mx = fmaxf(mx, lg[e]);
        float s = 0.f, pr[NE];
#pragma unroll
        for (int e = 0; e < NE; ++e) { pr[e] = __expf(lg[e] - mx); s += pr[e]; }
        float inv = 1.f / s;
#pragma unroll
        for (int e = 0; e < NE; ++e) probs[(size_t)t * NE + e] = pr[e] * inv;
        // top-2 with first-index tie break (matches jax.lax.top_k)
        int i0 = 0;
        for (int e = 1; e < NE; ++e) if (lg[e] > lg[i0]) i0 = e;
        int i1 = (i0 == 0) ? 1 : 0;
        for (int e = 0; e < NE; ++e)
            if (e != i0 && lg[e] > lg[i1]) i1 = e;
        float z  = __expf(lg[i1] - lg[i0]);
        float w0 = 1.f / (1.f + z);
        float w1 = z / (1.f + z);
        int p0 = atomicAdd(&count[i0], 1);
        toklist[i0 * TT + p0] = t; wlist[i0 * TT + p0] = w0;
        int p1 = atomicAdd(&count[i1], 1);
        toklist[i1 * TT + p1] = t; wlist[i1 * TT + p1] = w1;
    }
}

// ---------------- fused expert SwiGLU FFN (WMMA bf16, fp32 accum) ----------------
__global__ __launch_bounds__(256) void k_expert_ffn(
    const unsigned short* __restrict__ xbf,
    const unsigned short* __restrict__ w1bf,
    const unsigned short* __restrict__ w3bf,
    const unsigned short* __restrict__ w2bf,
    const int* __restrict__ toklist, const float* __restrict__ wlist,
    const int* __restrict__ count, float* __restrict__ out) {
    __shared__ __align__(16) unsigned short xs[16 * XS];
    __shared__ __align__(16) unsigned short ys[16 * YS];
    __shared__ int   stok[16];
    __shared__ float srw[16];

    const int e = blockIdx.y;
    const int tile = blockIdx.x;
    const int cnt = count[e];
    if (tile * 16 >= cnt) return;

    const int tid  = threadIdx.x;
    const int lane = tid & 31;
    const int wave = tid >> 5;
    const int mrow = lane & 15;               // row/col within 16x16 tile
    const int aoff = (lane < 16) ? 0 : 16;    // A fragment k base
    const int boff = (lane < 16) ? 0 : 16;    // B fragment k base

    if (tid < 16) {
        int g = tile * 16 + tid;
        if (g < cnt) { stok[tid] = toklist[e * TT + g]; srw[tid] = wlist[e * TT + g]; }
        else         { stok[tid] = 0;                   srw[tid] = 0.f; }
    }
    __syncthreads();

    // stage 16 gathered token rows (bf16) into LDS
    for (int idx = tid; idx < 16 * (DM / 8); idx += 256) {
        int r = idx >> 7, c = idx & 127;
        const v8bf* src = (const v8bf*)(xbf + (size_t)stok[r] * DM + c * 8);
        *(v8bf*)(&xs[r * XS + c * 8]) = *src;
    }
    __syncthreads();

    v8f acc[8];
#pragma unroll
    for (int i = 0; i < 8; ++i)
#pragma unroll
        for (int j = 0; j < 8; ++j) acc[i][j] = 0.f;

    for (int fb = 0; fb < FF / 128; ++fb) {
        // ---- phase 1: this wave computes y[:, fb*128 + wave*16 .. +16)
        const size_t w1base = ((size_t)e * FF + (size_t)(fb * 128 + wave * 16 + mrow)) * DM;
        const size_t w3base = w1base;  // same indexing into w3
        v8f hacc, vacc;
#pragma unroll
        for (int j = 0; j < 8; ++j) { hacc[j] = 0.f; vacc[j] = 0.f; }
#pragma unroll 4
        for (int kc = 0; kc < DM / 32; ++kc) {
            const int kk = kc * 32;
            v16bf a  = cat16(*(const v8bf*)&xs[mrow * XS + kk + aoff],
                             *(const v8bf*)&xs[mrow * XS + kk + aoff + 8]);
            v16bf b1 = cat16(*(const v8bf*)(w1bf + w1base + kk + boff),
                             *(const v8bf*)(w1bf + w1base + kk + boff + 8));
            v16bf b3 = cat16(*(const v8bf*)(w3bf + w3base + kk + boff),
                             *(const v8bf*)(w3bf + w3base + kk + boff + 8));
            hacc = __builtin_amdgcn_wmma_f32_16x16x32_bf16(false, a, false, b1,
                                                           (short)0, hacc, false, false);
            vacc = __builtin_amdgcn_wmma_f32_16x16x32_bf16(false, a, false, b3,
                                                           (short)0, vacc, false, false);
        }
        {
            const int rbase = (lane < 16) ? 0 : 8;
            const int ncol  = wave * 16 + mrow;
#pragma unroll
            for (int r = 0; r < 8; ++r) {
                float h = hacc[r], v = vacc[r];
                // silu(h)*v with 1-ulp v_rcp_f32 instead of the IEEE divide
                // expansion (bf16 inputs dominate the error budget anyway)
                float sig = __builtin_amdgcn_rcpf(1.f + __expf(-h));
                float y = h * v * sig;
                ys[(rbase + r) * YS + ncol] = f32_to_bf16_rne(y);
            }
        }
        __syncthreads();   // y chunk (16x128) complete

        // ---- phase 2: acc[:, wave*128 .. +128) += y(16x128) * w2^T chunk
#pragma unroll
        for (int kc = 0; kc < 4; ++kc) {
            const int kk = kc * 32;
            v16bf a = cat16(*(const v8bf*)&ys[mrow * YS + kk + aoff],
                            *(const v8bf*)&ys[mrow * YS + kk + aoff + 8]);
#pragma unroll
            for (int nt = 0; nt < 8; ++nt) {
                const size_t w2base =
                    ((size_t)e * DM + (size_t)(wave * 128 + nt * 16 + mrow)) * FF + fb * 128;
                v16bf b = cat16(*(const v8bf*)(w2bf + w2base + kk + boff),
                                *(const v8bf*)(w2bf + w2base + kk + boff + 8));
                acc[nt] = __builtin_amdgcn_wmma_f32_16x16x32_bf16(false, a, false, b,
                                                                  (short)0, acc[nt], false, false);
            }
        }
        __syncthreads();   // before next iteration overwrites ys
    }

    // scale rows by routing weight, scatter-add into output (2 adds/element total)
    const int rbase = (lane < 16) ? 0 : 8;
    float rw[8];
#pragma unroll
    for (int r = 0; r < 8; ++r) rw[r] = srw[rbase + r];
#pragma unroll
    for (int nt = 0; nt < 8; ++nt) {
        const int col = wave * 128 + nt * 16 + mrow;
#pragma unroll
        for (int r = 0; r < 8; ++r) {
            atomicAdd(&out[(size_t)stok[rbase + r] * DM + col], acc[nt][r] * rw[r]);
        }
    }
}

// ---------------- aux loss (deterministic tree reduction) ----------------
__global__ __launch_bounds__(256) void k_aux(const float* __restrict__ probs,
                                             const int* __restrict__ count,
                                             float* __restrict__ out_aux) {
    __shared__ float red[NE][256];
    const int tid = threadIdx.x;
    float p[NE];
#pragma unroll
    for (int e = 0; e < NE; ++e) p[e] = 0.f;
    for (int t = tid; t < TT; t += 256) {
#pragma unroll
        for (int e = 0; e < NE; ++e) p[e] += probs[(size_t)t * NE + e];
    }
#pragma unroll
    for (int e = 0; e < NE; ++e) red[e][tid] = p[e];
    __syncthreads();
    for (int off = 128; off > 0; off >>= 1) {
        if (tid < off) {
#pragma unroll
            for (int e = 0; e < NE; ++e) red[e][tid] += red[e][tid + off];
        }
        __syncthreads();
    }
    if (tid == 0) {
        float aux = 0.f;
#pragma unroll
        for (int e = 0; e < NE; ++e) {
            float ep = red[e][0] / (float)TT;
            float ef = (float)count[e] / (float)(TT * TOPK);
            aux += ep * ef;
        }
        out_aux[0] = aux * (float)NE;
    }
}

// ---------------- launch ----------------
extern "C" void kernel_launch(void* const* d_in, const int* in_sizes, int n_in,
                              void* d_out, int out_size, void* d_ws, size_t ws_size,
                              hipStream_t stream) {
    const float* x      = (const float*)d_in[0];
    const float* gate_w = (const float*)d_in[1];
    const float* w1     = (const float*)d_in[2];
    const float* w3     = (const float*)d_in[3];
    const float* w2     = (const float*)d_in[4];
    float* out = (float*)d_out;

    // workspace layout (~110 MB)
    char* ws = (char*)d_ws;
    size_t o = 0;
    unsigned short* xbf  = (unsigned short*)(ws + o); o += (size_t)TT * DM * 2;
    unsigned short* w1bf = (unsigned short*)(ws + o); o += (size_t)NE * FF * DM * 2;
    unsigned short* w3bf = (unsigned short*)(ws + o); o += (size_t)NE * FF * DM * 2;
    unsigned short* w2bf = (unsigned short*)(ws + o); o += (size_t)NE * DM * FF * 2;
    float* probs   = (float*)(ws + o); o += (size_t)TT * NE * 4;
    int*   toklist = (int*)(ws + o);   o += (size_t)NE * TT * 4;
    float* wlist   = (float*)(ws + o); o += (size_t)NE * TT * 4;
    int*   count   = (int*)(ws + o);   o += (size_t)NE * 4;

    hipMemsetAsync(d_out, 0, (size_t)out_size * sizeof(float), stream);
    hipMemsetAsync(count, 0, NE * sizeof(int), stream);

    k_f32_to_bf16<<<2048, 256, 0, stream>>>(x,  xbf,  TT * DM);
    k_f32_to_bf16<<<8192, 256, 0, stream>>>(w1, w1bf, NE * FF * DM);
    k_f32_to_bf16<<<8192, 256, 0, stream>>>(w3, w3bf, NE * FF * DM);
    k_f32_to_bf16<<<8192, 256, 0, stream>>>(w2, w2bf, NE * DM * FF);

    k_router<<<TT, 256, 0, stream>>>(x, gate_w, probs, toklist, wlist, count);

    dim3 grid(TT / 16, NE);   // worst case: one expert owns every token
    k_expert_ffn<<<grid, 256, 0, stream>>>(xbf, w1bf, w3bf, w2bf,
                                           toklist, wlist, count, out);

    k_aux<<<1, 256, 0, stream>>>(probs, count, out + (size_t)TT * DM);
}
